// MambaLayer_20753281974331
// MI455X (gfx1250) — matrix operand order repaired and checked by reading, not compiled, not run on hardware
//
#include <hip/hip_runtime.h>
#include <hip/hip_bf16.h>

#define S_LEN 2048
#define B_DIM 16
#define D_DIM 1024
#define N_DIM 1024

typedef __attribute__((ext_vector_type(16))) __bf16 v16bf;
typedef __attribute__((ext_vector_type(8)))  float  v8f;
typedef __attribute__((ext_vector_type(4)))  unsigned u32x4;
typedef __attribute__((ext_vector_type(8)))  int      i32x8;
typedef __attribute__((ext_vector_type(4)))  int      i32x4;

union Frag16 {
  v16bf v;
  unsigned short s[16];
  uint4 q[2];
};

__device__ __forceinline__ unsigned short f2bf_u(float f) {
  union { float f; unsigned u; } x; x.f = f;
  unsigned r = x.u + 0x7FFFu + ((x.u >> 16) & 1u);   // round-to-nearest-even
  return (unsigned short)(r >> 16);
}
__device__ __forceinline__ float bf2f(unsigned short s) {
  union { unsigned u; float f; } x; x.u = ((unsigned)s) << 16;
  return x.f;
}

// ---------------------------------------------------------------------------
// TDM: async 2D tile load global->LDS (Tensor Data Mover, D# per ISA 08).
// dims/strides in data_size units (bf16 -> code 1, 2 bytes).
// This toolchain exposes the 6-arg builtin:
//   (uint32x4 g0, int32x8 g1, int32x4 g2, int32x4 g3, int32x8 extra, i32 cpol)
// ---------------------------------------------------------------------------
__device__ __forceinline__ void tdm_load_tile_2d(unsigned short* dst_lds,
                                                 const unsigned short* gsrc,
                                                 unsigned tile0, unsigned tile1,
                                                 unsigned td0,
                                                 unsigned long long stride0) {
#if __has_builtin(__builtin_amdgcn_tensor_load_to_lds)
  const unsigned lds_off = (unsigned)(uintptr_t)dst_lds;  // flat LDS addr[31:0] = offset
  const unsigned long long ga = (unsigned long long)(uintptr_t)gsrc;
  const unsigned td1 = 1u << 20;                          // generous bound (OOB-safe)
  // D# group 0: count=1 | lds_addr | global_addr[56:0] | type=2
  u32x4 g0 = { 1u,
               lds_off,
               (unsigned)(ga & 0xFFFFFFFFu),
               (unsigned)(((ga >> 32) & 0x01FFFFFFu) | (2u << 30)) };
  // D# group 1: mask=0, data_size=1(2B), dims/tiles/stride packed per ISA layout
  i32x8 g1 = { (int)(1u << 16),
               (int)((td0 & 0xFFFFu) << 16),
               (int)(((td0 >> 16) & 0xFFFFu) | ((td1 & 0xFFFFu) << 16)),
               (int)(((td1 >> 16) & 0xFFFFu) | ((tile0 & 0xFFFFu) << 16)),
               (int)(tile1 & 0xFFFFu),
               (int)(stride0 & 0xFFFFFFFFu),
               (int)((stride0 >> 32) & 0xFFFFu),
               0 };
  i32x4 z4 = { 0, 0, 0, 0 };
  i32x8 z8 = { 0, 0, 0, 0, 0, 0, 0, 0 };
  __builtin_amdgcn_tensor_load_to_lds(g0, g1, z4, z4, z8, 0);
#else
  // fallback: synchronous copy by this wave
  const int lane = threadIdx.x & 31;
  for (unsigned r = 0; r < tile1; ++r)
    for (unsigned i = lane; i < tile0; i += 32)
      dst_lds[r * tile0 + i] = gsrc[r * stride0 + i];
#endif
}

__device__ __forceinline__ void tdm_wait() {
#if __has_builtin(__builtin_amdgcn_s_wait_tensorcnt)
  __builtin_amdgcn_s_wait_tensorcnt(0);
#endif
}

// ---------------------------------------------------------------------------
// fp32 -> bf16 streaming conversion (one compulsory pass; results L2-resident)
// ---------------------------------------------------------------------------
__global__ void __launch_bounds__(256) cvt_bf16_kernel(const float* __restrict__ src,
                                                       unsigned short* __restrict__ dst,
                                                       int n) {
  int i = blockIdx.x * 256 + threadIdx.x;
  if (i < n) dst[i] = f2bf_u(src[i]);
}

// ---------------------------------------------------------------------------
// Precompute: dt = softplus(x Wdt^T + bdt); a_bar = exp(dt*a);
//             b_bar = (x Wb^T + bb)*dt.
// Per wave: 16-row m-tile x 4 n-tiles, both projections fused (A reused 8x).
// All 8 B-fragments are loaded into distinct registers BEFORE the WMMA burst
// so loads of chunk k overlap the 8 WMMAs of chunk k (staged s_wait_loadcnt).
// ---------------------------------------------------------------------------
#define NT 4

__global__ void __launch_bounds__(256) precompute_kernel(
    const unsigned short* __restrict__ xbf,    // [S*B, D] bf16
    const unsigned short* __restrict__ wbbf,   // [N, D]  bf16
    const unsigned short* __restrict__ wdtbf,  // [N, D]  bf16
    const float* __restrict__ bb,
    const float* __restrict__ bdt,
    const float* __restrict__ lna,
    unsigned short* __restrict__ a_bar,        // [S*B, N] bf16
    unsigned short* __restrict__ b_bar)        // [S*B, N] bf16
{
  const int lane = threadIdx.x & 31;
  const int wave = threadIdx.x >> 5;
  const int gw   = blockIdx.x * 8 + wave;      // 32768 waves total
  const int tile_m = gw >> 4;                  // 2048 m-tiles
  const int ngrp   = gw & 15;                  // 16 groups of NT n-tiles
  const int lh = lane >> 4;
  const int c  = lane & 15;

  const int m0 = tile_m * 16;
  const int n0 = ngrp * (NT * 16);

  const unsigned short* arow = xbf + (size_t)(m0 + c) * D_DIM;

  v8f acc_dt[NT];
  v8f acc_b[NT];
#pragma unroll
  for (int j = 0; j < NT; ++j) { acc_dt[j] = {}; acc_b[j] = {}; }

  for (int k0 = 0; k0 < D_DIM; k0 += 32) {
    Frag16 a;
    a.q[0] = *(const uint4*)(arow + k0 + lh * 8);
    a.q[1] = *(const uint4*)(arow + k0 + lh * 8 + 16);

    Frag16 bd[NT], bw[NT];
#pragma unroll
    for (int j = 0; j < NT; ++j) {
      const size_t brow = (size_t)(n0 + j * 16 + c) * D_DIM + k0 + lh * 8;
      bd[j].q[0] = *(const uint4*)(wdtbf + brow);
      bd[j].q[1] = *(const uint4*)(wdtbf + brow + 16);
      bw[j].q[0] = *(const uint4*)(wbbf + brow);
      bw[j].q[1] = *(const uint4*)(wbbf + brow + 16);
    }
#pragma unroll
    for (int j = 0; j < NT; ++j) {
      acc_dt[j] = __builtin_amdgcn_wmma_f32_16x16x32_bf16(
          false, a.v, false, bd[j].v, (short)0, acc_dt[j], false, false);
      acc_b[j] = __builtin_amdgcn_wmma_f32_16x16x32_bf16(
          false, a.v, false, bw[j].v, (short)0, acc_b[j], false, false);
    }
  }

#pragma unroll
  for (int j = 0; j < NT; ++j) {
    const int ncol = n0 + j * 16 + c;
    const float bdt_n = bdt[ncol];
    const float bb_n  = bb[ncol];
    const float nega  = __expf(lna[ncol]);       // -a_n
#pragma unroll
    for (int vv = 0; vv < 8; ++vv) {
      const int m = m0 + vv + lh * 8;
      const float zdt = acc_dt[j][vv] + bdt_n;
      const float dt  = (zdt > 15.0f) ? zdt : __logf(1.0f + __expf(zdt));
      const float abv = __expf(-dt * nega);      // exp(dt * a)
      const float bbv = (acc_b[j][vv] + bb_n) * dt;
      const size_t idx = (size_t)m * N_DIM + ncol;
      a_bar[idx] = f2bf_u(abv);
      b_bar[idx] = f2bf_u(bbv);
    }
  }
}

// ---------------------------------------------------------------------------
// Persistent scan: 8 workgroups, each owns 128 output columns.
// Per step: TDM double-buffers a_bar/b_bar slices into LDS (issued during the
// previous step's WMMA phase, gated by s_wait_tensorcnt); h = a⊙y_prev + b is
// built in LDS; each wave runs a 16x16 bf16-WMMA tile over K=1024 against its
// L2-resident Wc slice; a device-scope atomic barrier serializes steps.
// ---------------------------------------------------------------------------
__global__ void __launch_bounds__(256) scan_kernel(
    const unsigned short* __restrict__ a_bar,
    const unsigned short* __restrict__ b_bar,
    const unsigned short* __restrict__ wcbf,   // [N, N] bf16 (B[k][n] = Wc[n][k])
    const float* __restrict__ bc,
    float* __restrict__ out,                   // [S, B, N] fp32
    int* __restrict__ cnt,                     // [S] zeroed counters
    int nwg)
{
  __shared__ unsigned short h_lds[B_DIM * N_DIM];        // 32 KB
  __shared__ unsigned short ab_lds[2][B_DIM * N_DIM];    // 64 KB (TDM staging)
  __shared__ unsigned short bb_lds[2][B_DIM * N_DIM];    // 64 KB

  const int tid  = threadIdx.x;
  const int lane = tid & 31;
  const int wave = tid >> 5;
  const int lh = lane >> 4;
  const int c  = lane & 15;
  const int BN = B_DIM * N_DIM;

  const int ncol = blockIdx.x * 128 + wave * 16 + c;
  const unsigned short* wrow = wcbf + (size_t)ncol * N_DIM;
  const float bc_n = bc[ncol];

  // prologue: kick off DMA of step 0 slices
  if (wave == 0) {
    tdm_load_tile_2d(ab_lds[0], a_bar, N_DIM, B_DIM, N_DIM, N_DIM);
    tdm_load_tile_2d(bb_lds[0], b_bar, N_DIM, B_DIM, N_DIM, N_DIM);
  }

  for (int t = 0; t < S_LEN; ++t) {
    const int cur = t & 1, nxt = cur ^ 1;
    const size_t base = (size_t)t * BN;
    const float* yprev = out + base - BN;       // dereferenced only for t >= 1

    if (wave == 0) tdm_wait();                  // step-t slices resident in LDS
    __syncthreads();

    // phase 1: h = a_bar ⊙ y_prev + b_bar (bf16 into LDS)
    if (t == 0) {
      for (int i = tid; i < BN; i += 256) h_lds[i] = bb_lds[0][i];
    } else {
      for (int i = tid; i < BN; i += 256) {
        float hv = fmaf(bf2f(ab_lds[cur][i]), yprev[i], bf2f(bb_lds[cur][i]));
        h_lds[i] = f2bf_u(hv);
      }
    }
    __syncthreads();

    // overlap next step's DMA with this step's WMMA chain
    if (t + 1 < S_LEN && wave == 0) {
      const size_t nb = base + BN;
      tdm_load_tile_2d(ab_lds[nxt], a_bar + nb, N_DIM, B_DIM, N_DIM, N_DIM);
      tdm_load_tile_2d(bb_lds[nxt], b_bar + nb, N_DIM, B_DIM, N_DIM, N_DIM);
    }

    // phase 2: y_tile = h[16xK] * WcT[Kx16], K = 1024 (unrolled x2 for overlap)
    v8f acc = {};
    for (int k0 = 0; k0 < N_DIM; k0 += 64) {
      Frag16 a0, b0, a1, b1;
      const unsigned short* ap0 = &h_lds[c * N_DIM + k0 + lh * 8];
      a0.q[0] = *(const uint4*)ap0;
      a0.q[1] = *(const uint4*)(ap0 + 16);
      a1.q[0] = *(const uint4*)(ap0 + 32);
      a1.q[1] = *(const uint4*)(ap0 + 48);
      const unsigned short* bp0 = wrow + k0 + lh * 8;
      b0.q[0] = *(const uint4*)bp0;
      b0.q[1] = *(const uint4*)(bp0 + 16);
      b1.q[0] = *(const uint4*)(bp0 + 32);
      b1.q[1] = *(const uint4*)(bp0 + 48);
      acc = __builtin_amdgcn_wmma_f32_16x16x32_bf16(
          false, a0.v, false, b0.v, (short)0, acc, false, false);
      acc = __builtin_amdgcn_wmma_f32_16x16x32_bf16(
          false, a1.v, false, b1.v, (short)0, acc, false, false);
    }

    // epilogue: write y_t (output doubles as the carry buffer)
#pragma unroll
    for (int vv = 0; vv < 8; ++vv) {
      const int m = vv + lh * 8;
      out[base + (size_t)m * N_DIM + ncol] = acc[vv] + bc_n;
    }

    __syncthreads();
    __threadfence();                       // release y_t writes
    if (tid == 0) {
      atomicAdd(&cnt[t], 1);
      while (__hip_atomic_load(&cnt[t], __ATOMIC_ACQUIRE,
                               __HIP_MEMORY_SCOPE_AGENT) < nwg) {
        __builtin_amdgcn_s_sleep(2);
      }
    }
    __syncthreads();
    __threadfence();                       // acquire other blocks' y_t
  }
}

// ---------------------------------------------------------------------------
// Launch. Workspace layout (~205 MB):
//   a_bar bf16 (67MB) | b_bar bf16 (67MB) | x bf16 (67MB) |
//   Wb bf16 (2MB) | Wdt bf16 (2MB) | Wc bf16 (2MB) | cnt (8KB)
// ---------------------------------------------------------------------------
extern "C" void kernel_launch(void* const* d_in, const int* in_sizes, int n_in,
                              void* d_out, int out_size, void* d_ws, size_t ws_size,
                              hipStream_t stream) {
  (void)in_sizes; (void)n_in; (void)out_size; (void)ws_size;
  const float* x   = (const float*)d_in[0];
  const float* Wb  = (const float*)d_in[1];
  const float* bb  = (const float*)d_in[2];
  const float* Wc  = (const float*)d_in[3];
  const float* bc  = (const float*)d_in[4];
  const float* Wdt = (const float*)d_in[5];
  const float* bdt = (const float*)d_in[6];
  const float* lna = (const float*)d_in[7];
  float* out = (float*)d_out;

  char* ws = (char*)d_ws;
  const size_t MROWS = (size_t)S_LEN * B_DIM;     // 32768
  const size_t AB    = MROWS * N_DIM;             // 33.5M elems
  size_t off = 0;
  unsigned short* a_bar = (unsigned short*)(ws + off); off += 2 * AB;
  unsigned short* b_bar = (unsigned short*)(ws + off); off += 2 * AB;
  unsigned short* xbf   = (unsigned short*)(ws + off); off += 2 * MROWS * D_DIM;
  unsigned short* wbbf  = (unsigned short*)(ws + off); off += 2 * (size_t)N_DIM * D_DIM;
  unsigned short* wdtbf = (unsigned short*)(ws + off); off += 2 * (size_t)N_DIM * D_DIM;
  unsigned short* wcbf  = (unsigned short*)(ws + off); off += 2 * (size_t)N_DIM * N_DIM;
  int* cnt = (int*)(ws + off);

  (void)hipMemsetAsync(cnt, 0, S_LEN * sizeof(int), stream);

  {
    int n = (int)(MROWS * D_DIM);
    cvt_bf16_kernel<<<(n + 255) / 256, 256, 0, stream>>>(x, xbf, n);
  }
  {
    int n = N_DIM * D_DIM;
    cvt_bf16_kernel<<<(n + 255) / 256, 256, 0, stream>>>(Wb, wbbf, n);
    cvt_bf16_kernel<<<(n + 255) / 256, 256, 0, stream>>>(Wdt, wdtbf, n);
  }
  {
    int n = N_DIM * N_DIM;
    cvt_bf16_kernel<<<(n + 255) / 256, 256, 0, stream>>>(Wc, wcbf, n);
  }

  // 2048 m-tiles * 16 n-groups = 32768 waves, 8 waves/block
  precompute_kernel<<<4096, 256, 0, stream>>>(xbf, wbbf, wdtbf, bb, bdt, lna,
                                              a_bar, b_bar);

  scan_kernel<<<8, 256, 0, stream>>>(a_bar, b_bar, wcbf, bc, out, cnt, 8);
}
